// GNN_auto_64149631533882
// MI455X (gfx1250) — matrix operand order repaired and checked by reading, not compiled, not run on hardware
//
#include <hip/hip_runtime.h>
#include <hip/hip_bf16.h>

// ---------------------------------------------------------------------------
// RED-GNN edge-attention message passing for MI455X (gfx1250, wave32, WMMA).
//
// pre  = relu([hs|hr|hqr] @ [Ws;Wr;Wqr] + b)        -> bf16 WMMA 16x16x32
// alpha= sigmoid(pre @ wa + b)                      -> bf16 WMMA vs col-0 frag
// agg[obj] += alpha * hs*hr  (f32 global atomics)
// out  = agg @ Wh                                   -> bf16 WMMA 16x16x32
// ---------------------------------------------------------------------------

typedef __attribute__((ext_vector_type(16))) __bf16 v16bf;
typedef __attribute__((ext_vector_type(8)))  __bf16 v8bf;
typedef __attribute__((ext_vector_type(8)))  float  v8f;

#define IN_DIM   128
#define KCAT     384          // 3*IN_DIM concatenated A width
#define NKC_CAT  12           // 384 / 32 k-chunks
#define NKC_ONE  4            // 128 / 32 k-chunks
#define NTILES   8            // 128 / 16 n-tiles

static __device__ __forceinline__ unsigned short f2bf(float f) {
    unsigned int u = __builtin_bit_cast(unsigned int, f);
    unsigned int r = u + 0x7FFFu + ((u >> 16) & 1u);   // round-to-nearest-even
    return (unsigned short)(r >> 16);
}

static __device__ __forceinline__ v8f zero8() {
    v8f z;
    #pragma unroll
    for (int i = 0; i < 8; ++i) z[i] = 0.0f;
    return z;
}

// A-fragment (bf16 16x32, wave32 layout): lane L holds row M = L%16.
// lanes 0-15 cover k = {0..15}, lanes 16-31 cover k = {16..31}.
// With row-major LDS rows (128 bf16 each) both 8-element halves are 16B-contiguous.
static __device__ __forceinline__ v16bf load_afrag(const unsigned short* row,
                                                   int kc, int lane) {
    int koff = kc * 32 + ((lane >> 4) & 1) * 16;
    v8bf a = *(const v8bf*)(row + koff);
    v8bf b = *(const v8bf*)(row + koff + 8);
    return __builtin_shufflevector(a, b, 0, 1, 2, 3, 4, 5, 6, 7,
                                         8, 9, 10, 11, 12, 13, 14, 15);
}

static __device__ __forceinline__ void lds_fence() {
    asm volatile("s_wait_dscnt 0x0" ::: "memory");
}

// ---------------------------------------------------------------------------
// Kernel 0: zero the f32 aggregation buffer (float4 stores).
// ---------------------------------------------------------------------------
__global__ void __launch_bounds__(256) zero_agg_kernel(float4* agg, long n4) {
    long i = (long)blockIdx.x * blockDim.x + threadIdx.x;
    if (i < n4) agg[i] = make_float4(0.f, 0.f, 0.f, 0.f);
}

// ---------------------------------------------------------------------------
// Kernel 1: pre-swizzle weights into bf16 B-fragment layout.
// B-fragment (bf16 32x16): lane L holds col n = L%16, k = (L<16?0:16)+j.
// Buffer layout: frag[(kc*NTILES + nt)*32 + lane][16] -> 32B contiguous / lane.
// ---------------------------------------------------------------------------
__global__ void __launch_bounds__(256) swizzle_weights_kernel(
    const float* __restrict__ Ws, const float* __restrict__ Wr,
    const float* __restrict__ Wqr, const float* __restrict__ wa_w,
    const float* __restrict__ Wh,
    unsigned short* __restrict__ wcat_frag,   // [12][8][32][16]
    unsigned short* __restrict__ wa_frag,     // [4][32][16] (col 0 only)
    unsigned short* __restrict__ wh_frag)     // [4][8][32][16]
{
    int t = blockIdx.x * blockDim.x + threadIdx.x;
    if (t < NKC_CAT * NTILES * 32) {                       // Wcat fragments
        int lane = t & 31, nt = (t >> 5) & 7, kc = t >> 8;
        int nn = lane & 15, kkb = (lane >> 4) * 16;
        unsigned short* dst = wcat_frag + t * 16;
        #pragma unroll
        for (int j = 0; j < 16; ++j) {
            int k = kc * 32 + kkb + j;                     // 0..383
            const float* W = (k < 128) ? Ws : (k < 256) ? Wr : Wqr;
            dst[j] = f2bf(W[(k & 127) * IN_DIM + nt * 16 + nn]);
        }
    } else if (t < 3072 + NKC_ONE * 32) {                  // wa_w column-0 frags
        int u = t - 3072;
        int lane = u & 31, kc = u >> 5;
        int nn = lane & 15, kkb = (lane >> 4) * 16;
        unsigned short* dst = wa_frag + u * 16;
        #pragma unroll
        for (int j = 0; j < 16; ++j)
            dst[j] = (nn == 0) ? f2bf(wa_w[kc * 32 + kkb + j]) : (unsigned short)0;
    } else if (t < 3072 + 128 + NKC_ONE * NTILES * 32) {   // Wh fragments
        int u = t - 3200;
        int lane = u & 31, nt = (u >> 5) & 7, kc = u >> 8;
        int nn = lane & 15, kkb = (lane >> 4) * 16;
        unsigned short* dst = wh_frag + u * 16;
        #pragma unroll
        for (int j = 0; j < 16; ++j)
            dst[j] = f2bf(Wh[(kc * 32 + kkb + j) * IN_DIM + nt * 16 + nn]);
    }
}

// ---------------------------------------------------------------------------
// Kernel 2: edge attention + gated message scatter. 1 wave == 16 edges.
// ---------------------------------------------------------------------------
__global__ void __launch_bounds__(128) edge_attn_kernel(
    const int* __restrict__ q_rel, const int* __restrict__ r_idx,
    const float* __restrict__ hidden,
    const int* __restrict__ edge_sub, const int* __restrict__ edge_rel,
    const int* __restrict__ edge_obj,
    const float* __restrict__ rela,
    const float* __restrict__ Wqr_b, const float* __restrict__ wa_b_p,
    const unsigned short* __restrict__ wcat_frag,
    const unsigned short* __restrict__ wa_frag,
    float* __restrict__ agg, int E)
{
    __shared__ unsigned short stage[4][3][16][IN_DIM];   // bf16 [hs|hr|hqr] tiles
    __shared__ float alphas[4][16];

    const int wave = threadIdx.x >> 5;
    const int lane = threadIdx.x & 31;
    const long tile = (long)blockIdx.x * 4 + wave;
    const long base = tile * 16;
    if (base >= E) return;                                // uniform per wave

    // ---- gather + f32->bf16 stage: lane pair per edge row --------------------
    const int  r     = lane >> 1;
    const int  cbase = (lane & 1) * 64;
    const long e     = base + r;
    const bool ev    = (e < E);
    const int  sub   = ev ? edge_sub[e] : 0;
    const int  rel   = ev ? edge_rel[e] : 0;
    const int  qr    = ev ? q_rel[r_idx[e]] : 0;
    const float* hsrow = hidden + (long)sub * IN_DIM;
    const float* hrrow = rela   + (long)rel * IN_DIM;
    const float* hqrow = rela   + (long)qr  * IN_DIM;

    {
        const float* srcs[3] = {hsrow, hrrow, hqrow};
        #pragma unroll
        for (int s = 0; s < 3; ++s) {
            unsigned short* dst = &stage[wave][s][r][0];
            #pragma unroll
            for (int c = 0; c < 64; c += 4) {
                float4 v = *(const float4*)(srcs[s] + cbase + c);
                *(unsigned int*)(dst + cbase + c) =
                    (unsigned int)f2bf(v.x) | ((unsigned int)f2bf(v.y) << 16);
                *(unsigned int*)(dst + cbase + c + 2) =
                    (unsigned int)f2bf(v.z) | ((unsigned int)f2bf(v.w) << 16);
            }
        }
    }
    lds_fence();

    // ---- pre = Acat(16x384) @ Wcat(384x128): 96 bf16 WMMAs -------------------
    v8f acc[NTILES];
    #pragma unroll
    for (int nt = 0; nt < NTILES; ++nt) acc[nt] = zero8();

    const int m = lane & 15;
    #pragma unroll
    for (int kc = 0; kc < NKC_CAT; ++kc) {
        const unsigned short* rowp = &stage[wave][kc >> 2][m][0];
        v16bf afrag = load_afrag(rowp, kc & 3, lane);
        #pragma unroll
        for (int nt = 0; nt < NTILES; ++nt) {
            v16bf bfrag = *(const v16bf*)(wcat_frag + ((kc * NTILES + nt) * 32 + lane) * 16);
            acc[nt] = __builtin_amdgcn_wmma_f32_16x16x32_bf16(
                false, afrag, false, bfrag, (short)0, acc[nt], false, false);
        }
    }

    // ---- bias + relu, write pre (bf16) back to LDS (reuse hs slab) -----------
    const int ncol = lane & 15;
    const int mhi  = (lane >> 4) << 3;                    // +8 for lanes 16-31
    unsigned short* prep = &stage[wave][0][0][0];
    #pragma unroll
    for (int nt = 0; nt < NTILES; ++nt) {
        float b = Wqr_b[nt * 16 + ncol];
        #pragma unroll
        for (int v = 0; v < 8; ++v) {
            float x = acc[nt][v] + b;
            x = x > 0.f ? x : 0.f;
            prep[(v + mhi) * IN_DIM + nt * 16 + ncol] = f2bf(x);
        }
    }
    lds_fence();

    // ---- alpha = sigmoid(pre @ wa + b): 4 WMMAs, result in column 0 ----------
    v8f accw = zero8();
    #pragma unroll
    for (int kc = 0; kc < NKC_ONE; ++kc) {
        v16bf afrag = load_afrag(&stage[wave][0][m][0], kc, lane);
        v16bf bfrag = *(const v16bf*)(wa_frag + (kc * 32 + lane) * 16);
        accw = __builtin_amdgcn_wmma_f32_16x16x32_bf16(
            false, afrag, false, bfrag, (short)0, accw, false, false);
    }
    const float wab = wa_b_p[0];
    if ((lane & 15) == 0) {                               // col 0 lanes: 0 and 16
        #pragma unroll
        for (int v = 0; v < 8; ++v) {
            float x = accw[v] + wab;
            alphas[wave][v + mhi] = 1.0f / (1.0f + __expf(-x));
        }
    }
    lds_fence();

    // ---- scatter: agg[obj] += alpha * hs * hr (f32, global atomics) ----------
    if (ev) {
        const float al  = alphas[wave][r];
        const int   obj = edge_obj[e];
        float* dst = agg + (long)obj * IN_DIM + cbase;
        #pragma unroll
        for (int c = 0; c < 64; c += 4) {
            float4 a4 = *(const float4*)(hsrow + cbase + c);
            float4 b4 = *(const float4*)(hrrow + cbase + c);
            unsafeAtomicAdd(dst + c + 0, al * a4.x * b4.x);
            unsafeAtomicAdd(dst + c + 1, al * a4.y * b4.y);
            unsafeAtomicAdd(dst + c + 2, al * a4.z * b4.z);
            unsafeAtomicAdd(dst + c + 3, al * a4.w * b4.w);
        }
    }
}

// ---------------------------------------------------------------------------
// Kernel 3: out = agg(50000x128) @ Wh(128x128). 1 wave == 16 rows.
// ---------------------------------------------------------------------------
__global__ void __launch_bounds__(128) out_gemm_kernel(
    const float* __restrict__ agg, const unsigned short* __restrict__ wh_frag,
    float* __restrict__ out, int nnode)
{
    __shared__ unsigned short stage[4][16][IN_DIM];

    const int wave = threadIdx.x >> 5;
    const int lane = threadIdx.x & 31;
    const long tile = (long)blockIdx.x * 4 + wave;
    const long base = tile * 16;
    if (base >= nnode) return;

    const int r     = lane >> 1;
    const int cbase = (lane & 1) * 64;
    const long row  = base + r;
    const bool rv   = (row < nnode);
    const float* src = agg + (rv ? row : 0) * (long)IN_DIM;
    unsigned short* dstld = &stage[wave][r][0];
    #pragma unroll
    for (int c = 0; c < 64; c += 4) {
        float4 v = *(const float4*)(src + cbase + c);
        if (!rv) v = make_float4(0.f, 0.f, 0.f, 0.f);
        *(unsigned int*)(dstld + cbase + c) =
            (unsigned int)f2bf(v.x) | ((unsigned int)f2bf(v.y) << 16);
        *(unsigned int*)(dstld + cbase + c + 2) =
            (unsigned int)f2bf(v.z) | ((unsigned int)f2bf(v.w) << 16);
    }
    lds_fence();

    const int m    = lane & 15;
    const int ncol = lane & 15;
    const int mhi  = (lane >> 4) << 3;
    #pragma unroll
    for (int nt = 0; nt < NTILES; ++nt) {
        v8f acc = zero8();
        #pragma unroll
        for (int kc = 0; kc < NKC_ONE; ++kc) {
            v16bf afrag = load_afrag(&stage[wave][m][0], kc, lane);
            v16bf bfrag = *(const v16bf*)(wh_frag + ((kc * NTILES + nt) * 32 + lane) * 16);
            acc = __builtin_amdgcn_wmma_f32_16x16x32_bf16(
                false, afrag, false, bfrag, (short)0, acc, false, false);
        }
        #pragma unroll
        for (int v = 0; v < 8; ++v) {
            long rr = base + v + mhi;
            if (rr < nnode) out[rr * IN_DIM + nt * 16 + ncol] = acc[v];
        }
    }
}

// ---------------------------------------------------------------------------
extern "C" void kernel_launch(void* const* d_in, const int* in_sizes, int n_in,
                              void* d_out, int out_size, void* d_ws, size_t ws_size,
                              hipStream_t stream) {
    const int*   q_rel    = (const int*)d_in[1];
    const int*   r_idx    = (const int*)d_in[2];
    const float* hidden   = (const float*)d_in[3];
    const int*   edge_sub = (const int*)d_in[4];
    const int*   edge_rel = (const int*)d_in[5];
    const int*   edge_obj = (const int*)d_in[6];
    const float* rela     = (const float*)d_in[8];
    const float* Ws       = (const float*)d_in[9];
    const float* Wr       = (const float*)d_in[10];
    const float* Wqr_w    = (const float*)d_in[11];
    const float* Wqr_b    = (const float*)d_in[12];
    const float* wa_w     = (const float*)d_in[13];
    const float* wa_b     = (const float*)d_in[14];
    const float* Wh       = (const float*)d_in[15];

    const int E     = in_sizes[4];
    const int nnode = in_sizes[3] / IN_DIM;

    // d_ws layout: [agg f32 nnode*128] [wcat frags] [wa frags] [wh frags]
    char* ws = (char*)d_ws;
    size_t agg_bytes = (size_t)nnode * IN_DIM * sizeof(float);
    agg_bytes = (agg_bytes + 255) & ~(size_t)255;
    float*          agg   = (float*)ws;
    unsigned short* wcat  = (unsigned short*)(ws + agg_bytes);                 // 98304 B
    unsigned short* wafr  = (unsigned short*)(ws + agg_bytes + 98304);         //  4096 B
    unsigned short* whfr  = (unsigned short*)(ws + agg_bytes + 98304 + 4096);  // 32768 B

    // 0) zero accumulator
    long n4 = (long)nnode * IN_DIM / 4;
    zero_agg_kernel<<<(unsigned)((n4 + 255) / 256), 256, 0, stream>>>((float4*)agg, n4);

    // 1) swizzle weights into WMMA B-fragment layout (bf16)
    swizzle_weights_kernel<<<17, 256, 0, stream>>>(Ws, Wr, Wqr_w, wa_w, Wh,
                                                   wcat, wafr, whfr);

    // 2) per-edge attention + gated scatter (16 edges per wave, 4 waves/block)
    long etiles = (E + 15) / 16;
    edge_attn_kernel<<<(unsigned)((etiles + 3) / 4), 128, 0, stream>>>(
        q_rel, r_idx, hidden, edge_sub, edge_rel, edge_obj, rela,
        Wqr_b, wa_b, wcat, wafr, agg, E);

    // 3) out = agg @ Wh
    long rtiles = (nnode + 15) / 16;
    out_gemm_kernel<<<(unsigned)((rtiles + 3) / 4), 128, 0, stream>>>(
        agg, whfr, (float*)d_out, nnode);
}
